// HybridHymbaBlockCPU_5136780886733
// MI455X (gfx1250) — matrix operand order repaired and checked by reading, not compiled, not run on hardware
//
#include <hip/hip_runtime.h>
#include <hip/hip_bf16.h>

// ---------------- problem constants ----------------
#define B_    4
#define T_    2048
#define BT_   8192          // B*T
#define D_    256
#define H_    8
#define DH_   32
#define D3_   85            // D // 3
#define MD_   768           // MULT * D
#define R_    8
#define RANK_ 16
#define SCALE_ 1.5f

typedef __bf16 bf16_t;
typedef __attribute__((ext_vector_type(16))) __bf16 v16bf;
typedef __attribute__((ext_vector_type(8)))  __bf16 v8bf;
typedef __attribute__((ext_vector_type(8)))  float  v8f;
typedef __attribute__((ext_vector_type(4)))  unsigned int u32x4;
typedef __attribute__((ext_vector_type(8)))  int i32x8;
typedef __attribute__((ext_vector_type(4)))  int i32x4;

#define IMIN(a,b) ((a)<(b)?(a):(b))

__device__ __forceinline__ v16bf comb(v8bf lo, v8bf hi) {
  v16bf r;
#pragma unroll
  for (int i = 0; i < 8; ++i) { r[i] = lo[i]; r[i + 8] = hi[i]; }
  return r;
}
__device__ __forceinline__ float gelu_f(float x) {
  return 0.5f * x * (1.0f + erff(x * 0.70710678118654752f));
}
__device__ __forceinline__ float sigmoid_f(float x) {
  return 1.0f / (1.0f + __expf(-x));
}

// ---------------- TDM: 2-D tile (tile_rows x row_dwords DWORDS) global -> LDS ----------
// Descriptor per CDNA5 ISA ch.8. Rows past rows_avail read as zero (OOB) -> LDS
// tile is zero-padded for free. stride_dw = memory row pitch in dwords.
__device__ __forceinline__ void tdm_load_2d(const void* gptr, unsigned lds_off,
                                            unsigned row_dwords, unsigned rows_avail,
                                            unsigned tile_rows, unsigned stride_dw)
{
  unsigned long long ga = (unsigned long long)(uintptr_t)gptr;
  u32x4 g0;
  g0[0] = 1u;                                               // count=1, user mode
  g0[1] = lds_off;                                          // lds_addr (bytes)
  g0[2] = (unsigned)(ga & 0xFFFFFFFFu);                     // global_addr[31:0]
  g0[3] = (unsigned)((ga >> 32) & 0x01FFFFFFu) | (2u << 30);// global_addr[56:32] | type=2
  i32x8 g1;
  g1[0] = 0x00020000;                                       // wg_mask=0, data_size=2 (4B)
  g1[1] = (int)((row_dwords & 0xFFFFu) << 16);              // tensor_dim0[15:0]
  g1[2] = (int)((row_dwords >> 16) | ((rows_avail & 0xFFFFu) << 16)); // dim0 hi | tensor_dim1 lo
  g1[3] = (int)((rows_avail >> 16) | (row_dwords << 16));   // tensor_dim1 hi | tile_dim0
  g1[4] = (int)(tile_rows & 0xFFFFu);                       // tile_dim1, tile_dim2=0
  g1[5] = (int)stride_dw;                                   // tensor_dim0_stride[31:0]
  g1[6] = 0;
  g1[7] = 0;
  i32x4 z4 = {0, 0, 0, 0};
#if defined(__clang_major__) && (__clang_major__ >= 23)
  i32x8 z8 = {0, 0, 0, 0, 0, 0, 0, 0};
  __builtin_amdgcn_tensor_load_to_lds(g0, g1, z4, z4, z8, 0);
#else
  __builtin_amdgcn_tensor_load_to_lds(g0, g1, z4, z4, 0);
#endif
}

// ---------------- weight prep ----------------
__global__ __launch_bounds__(256) void fuse_lora_kernel(
    const float* __restrict__ W, const float* __restrict__ A,
    const float* __restrict__ Bm, bf16_t* __restrict__ out, int N, int K)
{
  size_t idx = (size_t)blockIdx.x * 256 + threadIdx.x;
  if (idx >= (size_t)N * K) return;
  int n = (int)(idx / K), k = (int)(idx % K);
  float s = W[idx];
#pragma unroll
  for (int r = 0; r < R_; ++r) s += SCALE_ * A[n * R_ + r] * Bm[(size_t)r * K + k];
  out[idx] = (bf16_t)s;
}

__global__ __launch_bounds__(256) void cvt_bf16_kernel(
    const float* __restrict__ src, bf16_t* __restrict__ dst, int n)
{
  int i = blockIdx.x * 256 + threadIdx.x;
  if (i < n) dst[i] = (bf16_t)src[i];
}

// src [N,Ks] f32 -> dst [N,Kd] bf16, zero-padded cols Ks..Kd-1
__global__ __launch_bounds__(256) void cvt_pad_kernel(
    const float* __restrict__ src, bf16_t* __restrict__ dst, int N, int Ks, int Kd)
{
  int i = blockIdx.x * 256 + threadIdx.x;
  if (i >= N * Kd) return;
  int n = i / Kd, k = i % Kd;
  dst[i] = (k < Ks) ? (bf16_t)src[n * Ks + k] : (bf16_t)0.0f;
}

// ---------------- layernorm (one block per row, D=256) ----------------
__global__ __launch_bounds__(256) void ln_kernel(
    const float* __restrict__ x, const float* __restrict__ g,
    const float* __restrict__ b, bf16_t* __restrict__ out)
{
  __shared__ float red[256];
  const int row = blockIdx.x, tid = threadIdx.x;
  float v = x[(size_t)row * D_ + tid];
  red[tid] = v; __syncthreads();
  for (int s = 128; s > 0; s >>= 1) { if (tid < s) red[tid] += red[tid + s]; __syncthreads(); }
  float mean = red[0] * (1.0f / D_); __syncthreads();
  float d = v - mean;
  red[tid] = d * d; __syncthreads();
  for (int s = 128; s > 0; s >>= 1) { if (tid < s) red[tid] += red[tid + s]; __syncthreads(); }
  float var = red[0] * (1.0f / D_);
  float y = d * rsqrtf(var + 1e-5f) * g[tid] + b[tid];
  out[(size_t)row * D_ + tid] = (bf16_t)y;
}

// ---------------- generic bf16 WMMA GEMM, weights double-buffered via TDM -------------
// C[M,N] = act(A[M,K] @ Wt[Nw,K]^T + bias) (+resid). N is the output stride.
// Requires K % 32 == 0, M % 128 == 0, kslab = min(K,256), K % kslab == 0.
// Block: 256 thr = 8 waves, tile 128x64, wave tile 32x32 (2x2 WMMA 16x16x32).
// Weight K-slabs are DMA'd by the TDM into ping-pong LDS buffers; slab s+1 is in
// flight while slab s feeds the WMMAs (TDM completes in-order per wave, so
// s_wait_tensorcnt 1 guarantees slab s while s+1 is still outstanding).
template <bool NGUARD>
__global__ __launch_bounds__(256) void gemm_bf16_kernel(
    const bf16_t* __restrict__ A, const bf16_t* __restrict__ Wt,
    const float* __restrict__ bias, const float* __restrict__ resid,
    float* __restrict__ C, bf16_t* __restrict__ Cb,
    int M, int N, int K, int Nw, int act, int kslab)
{
  extern __shared__ __align__(16) unsigned char smem_raw[];
  bf16_t* buf0 = (bf16_t*)smem_raw;                  // 2 x [64][kslab] bf16
  const int lane = threadIdx.x & 31;
  const int wid  = threadIdx.x >> 5;
  const int half = lane >> 4;
  const int l16  = lane & 15;
  const int n0 = blockIdx.y * 64;
  const int wm = blockIdx.x * 128 + (wid >> 1) * 32;
  const int wnl = (wid & 1) * 32;                    // col base within the LDS tile
  const int nslab = K / kslab;
  const unsigned slab_bytes = (unsigned)(64 * kslab * 2);

  if (wid == 0)
    tdm_load_2d(Wt + (size_t)n0 * K, 0u, (unsigned)(kslab >> 1),
                (unsigned)(Nw - n0), 64u, (unsigned)(K >> 1));

  const int akb = half * 8;    // A-frag K chunks: [akb..akb+7], [akb+16..akb+23]
  const int bkb = half * 8;    // B-frag K chunks: [bkb..bkb+7], [bkb+16..bkb+23]
  v8f acc[2][2] = {};

  for (int s = 0; s < nslab; ++s) {
    if (wid == 0) {
      if (s + 1 < nslab) {
        tdm_load_2d(Wt + (size_t)n0 * K + (size_t)(s + 1) * kslab,
                    (unsigned)((s + 1) & 1) * slab_bytes, (unsigned)(kslab >> 1),
                    (unsigned)(Nw - n0), 64u, (unsigned)(K >> 1));
        __builtin_amdgcn_s_wait_tensorcnt(1);   // slab s complete, s+1 in flight
      } else {
        __builtin_amdgcn_s_wait_tensorcnt(0);   // last slab complete
      }
    }
    __syncthreads();

    const bf16_t* wtile = buf0 + (size_t)(s & 1) * 64 * kslab;
    const bf16_t* b0 = wtile + (size_t)(wnl + l16) * kslab;
    const bf16_t* b1 = wtile + (size_t)(wnl + 16 + l16) * kslab;
    const bf16_t* a0 = A + (size_t)(wm + l16) * K + (size_t)s * kslab;
    const bf16_t* a1 = A + (size_t)(wm + 16 + l16) * K + (size_t)s * kslab;
    __builtin_prefetch(a0 + kslab, 0, 1);
    __builtin_prefetch(a1 + kslab, 0, 1);

    for (int k0 = 0; k0 < kslab; k0 += 32) {
      v16bf af0 = comb(*(const v8bf*)(a0 + k0 + akb), *(const v8bf*)(a0 + k0 + akb + 16));
      v16bf af1 = comb(*(const v8bf*)(a1 + k0 + akb), *(const v8bf*)(a1 + k0 + akb + 16));
      v16bf bt0 = comb(*(const v8bf*)(b0 + k0 + bkb), *(const v8bf*)(b0 + k0 + bkb + 16));
      v16bf bt1 = comb(*(const v8bf*)(b1 + k0 + bkb), *(const v8bf*)(b1 + k0 + bkb + 16));
      acc[0][0] = __builtin_amdgcn_wmma_f32_16x16x32_bf16(false, af0, false, bt0, (short)0, acc[0][0], false, false);
      acc[0][1] = __builtin_amdgcn_wmma_f32_16x16x32_bf16(false, af0, false, bt1, (short)0, acc[0][1], false, false);
      acc[1][0] = __builtin_amdgcn_wmma_f32_16x16x32_bf16(false, af1, false, bt0, (short)0, acc[1][0], false, false);
      acc[1][1] = __builtin_amdgcn_wmma_f32_16x16x32_bf16(false, af1, false, bt1, (short)0, acc[1][1], false, false);
    }
    __syncthreads();   // readers done before TDM reuses this buffer
  }

#pragma unroll
  for (int mt = 0; mt < 2; ++mt)
#pragma unroll
    for (int nt = 0; nt < 2; ++nt) {
      int col = n0 + wnl + 16 * nt + l16;
      if (NGUARD && col >= N) continue;
      float bv = bias ? bias[col] : 0.0f;
#pragma unroll
      for (int r = 0; r < 8; ++r) {
        int row = wm + 16 * mt + half * 8 + r;
        float v = acc[mt][nt][r] + bv;
        if (act == 1)      v = gelu_f(v);
        else if (act == 2) v = sigmoid_f(v);
        size_t idx = (size_t)row * N + col;
        if (resid) v += resid[idx];
        if (C)  C[idx]  = v;
        if (Cb) Cb[idx] = (bf16_t)v;
      }
    }
}

// ---------------- RoPE split + V transpose ----------------
__global__ __launch_bounds__(256) void rope_kernel(
    const float* __restrict__ qkv, bf16_t* __restrict__ Qb,
    bf16_t* __restrict__ Kb, bf16_t* __restrict__ Vt)
{
  size_t gid = (size_t)blockIdx.x * 256 + threadIdx.x;  // B*T*H*16 threads
  int i = (int)(gid & 15);
  int h = (int)((gid >> 4) & 7);
  size_t bt = gid >> 7;
  int t = (int)(bt & (T_ - 1));
  int b = (int)(bt >> 11);
  const float* row = qkv + bt * (3 * D_);
  float theta = __powf(10000.0f, -(float)(2 * i) / (float)DH_);
  float sn, cs; __sincosf((float)t * theta, &sn, &cs);
  float q1 = row[h * DH_ + i],            q2 = row[h * DH_ + i + 16];
  float k1 = row[D_ + h * DH_ + i],       k2 = row[D_ + h * DH_ + i + 16];
  float v1 = row[2 * D_ + h * DH_ + i],   v2 = row[2 * D_ + h * DH_ + i + 16];
  size_t bh = (size_t)b * H_ + h;
  bf16_t* qp = Qb + (bh * T_ + t) * DH_;
  bf16_t* kp = Kb + (bh * T_ + t) * DH_;
  qp[i]      = (bf16_t)(q1 * cs - q2 * sn);
  qp[i + 16] = (bf16_t)(q1 * sn + q2 * cs);
  kp[i]      = (bf16_t)(k1 * cs - k2 * sn);
  kp[i + 16] = (bf16_t)(k1 * sn + k2 * cs);
  Vt[(bh * DH_ + i) * T_ + t]      = (bf16_t)v1;
  Vt[(bh * DH_ + i + 16) * T_ + t] = (bf16_t)v2;
}

// ---------------- causal flash attention (WMMA) ----------------
__global__ __launch_bounds__(128) void attn_kernel(
    const bf16_t* __restrict__ Q, const bf16_t* __restrict__ Kmat,
    const bf16_t* __restrict__ Vt, bf16_t* __restrict__ Zb)
{
  __shared__ __align__(16) bf16_t pbuf[4][16 * 32];
  const int lane = threadIdx.x & 31;
  const int wid  = threadIdx.x >> 5;
  const int half = lane >> 4;
  const int l16  = lane & 15;
  const int bh = blockIdx.y;
  const int b = bh >> 3, h = bh & 7;
  const int q0 = blockIdx.x * 64 + wid * 16;
  const bf16_t* Qp = Q    + (size_t)bh * T_ * DH_;
  const bf16_t* Kp = Kmat + (size_t)bh * T_ * DH_;
  const bf16_t* Vp = Vt   + (size_t)bh * DH_ * T_;

  const bf16_t* qrowp = Qp + (size_t)(q0 + l16) * DH_;
  v16bf qf = comb(*(const v8bf*)(qrowp + half * 16),
                  *(const v8bf*)(qrowp + half * 16 + 8));

  v8f accv[2] = {};
  float mrow[8], lrow[8];
#pragma unroll
  for (int r = 0; r < 8; ++r) { mrow[r] = -1e30f; lrow[r] = 0.0f; }
  const float scl = 0.17677669529663687f;  // 1/sqrt(32)
  const int kend = q0 + 16;

  for (int s0 = 0; s0 < kend; s0 += 32) {
    int kr0 = s0 + l16;
    int kr1 = s0 + 16 + l16;
    const bf16_t* kp0 = Kp + (size_t)IMIN(kr0, T_ - 1) * DH_ + half * 16;
    const bf16_t* kp1 = Kp + (size_t)IMIN(kr1, T_ - 1) * DH_ + half * 16;
    v16bf kf0 = comb(*(const v8bf*)kp0, *(const v8bf*)(kp0 + 8));
    v16bf kf1 = comb(*(const v8bf*)kp1, *(const v8bf*)(kp1 + 8));
    v8f zf = {};
    v8f s0t = __builtin_amdgcn_wmma_f32_16x16x32_bf16(false, qf, false, kf0, (short)0, zf, false, false);
    v8f s1t = __builtin_amdgcn_wmma_f32_16x16x32_bf16(false, qf, false, kf1, (short)0, zf, false, false);

#pragma unroll
    for (int r = 0; r < 8; ++r) {
      int qrow = q0 + half * 8 + r;
      float e0 = (kr0 <= qrow) ? s0t[r] * scl : -1e30f;
      float e1 = (kr1 <= qrow) ? s1t[r] * scl : -1e30f;
      float mx = fmaxf(e0, e1);
#pragma unroll
      for (int off = 1; off < 16; off <<= 1) mx = fmaxf(mx, __shfl_xor(mx, off, 32));
      float mnew = fmaxf(mrow[r], mx);
      float corr = __expf(mrow[r] - mnew);
      float p0 = __expf(e0 - mnew);
      float p1 = __expf(e1 - mnew);
      float rs = p0 + p1;
#pragma unroll
      for (int off = 1; off < 16; off <<= 1) rs += __shfl_xor(rs, off, 32);
      lrow[r] = lrow[r] * corr + rs;
      mrow[r] = mnew;
      accv[0][r] *= corr;
      accv[1][r] *= corr;
      pbuf[wid][(half * 8 + r) * 32 + l16]      = (bf16_t)p0;
      pbuf[wid][(half * 8 + r) * 32 + 16 + l16] = (bf16_t)p1;
    }
    asm volatile("s_wait_dscnt 0" ::: "memory");  // LDS RAW turnaround within wave

    const bf16_t* pr = &pbuf[wid][l16 * 32];
    v16bf pf = comb(*(const v8bf*)(pr + half * 16),
                    *(const v8bf*)(pr + half * 16 + 8));
    const bf16_t* vp0 = Vp + (size_t)l16 * T_ + s0 + half * 16;
    const bf16_t* vp1 = Vp + (size_t)(16 + l16) * T_ + s0 + half * 16;
    v16bf vf0 = comb(*(const v8bf*)vp0, *(const v8bf*)(vp0 + 8));
    v16bf vf1 = comb(*(const v8bf*)vp1, *(const v8bf*)(vp1 + 8));
    accv[0] = __builtin_amdgcn_wmma_f32_16x16x32_bf16(false, pf, false, vf0, (short)0, accv[0], false, false);
    accv[1] = __builtin_amdgcn_wmma_f32_16x16x32_bf16(false, pf, false, vf1, (short)0, accv[1], false, false);
  }

#pragma unroll
  for (int nt = 0; nt < 2; ++nt)
#pragma unroll
    for (int r = 0; r < 8; ++r) {
      int t = q0 + half * 8 + r;
      float ov = accv[nt][r] / lrow[r];
      Zb[((size_t)b * T_ + t) * D_ + h * DH_ + nt * 16 + l16] = (bf16_t)ov;
    }
}

// ---------------- alpha / lambda gating ----------------
__global__ __launch_bounds__(256) void alpha_lam_kernel(
    const float* __restrict__ e1, const float* __restrict__ E2w,
    const float* __restrict__ E2b, const float* __restrict__ a_base,
    const float* __restrict__ a_delta, float* __restrict__ lam)
{
  __shared__ float es[D3_];
  const int bt = blockIdx.x, d = threadIdx.x;
  if (d < D3_) es[d] = e1[(size_t)bt * D3_ + d];
  __syncthreads();
  float s0 = E2b[0], s1 = E2b[1], s2 = E2b[2];
  for (int j = 0; j < D3_; ++j) {
    float e = es[j];
    s0 += e * E2w[j];
    s1 += e * E2w[D3_ + j];
    s2 += e * E2w[2 * D3_ + j];
  }
  float al = (d < D3_) ? sigmoid_f(s0) : (d < 2 * D3_) ? sigmoid_f(s1) : sigmoid_f(s2);
  float ae = a_base[d] + al * tanhf(a_delta[d]);
  lam[(size_t)bt * D_ + d] = sigmoid_f(ae);  // exp(-softplus(-a)) == sigmoid(a)
}

// ---------------- sequential scan over T ----------------
__global__ __launch_bounds__(256) void scan_kernel(
    const float* __restrict__ lam, const float* __restrict__ Bu,
    bf16_t* __restrict__ Sb)
{
  int g = blockIdx.x * 256 + threadIdx.x;  // B*D threads
  int b = g >> 8, d = g & 255;
  float s = 0.0f;
  for (int t = 0; t < T_; ++t) {
    size_t idx = ((size_t)b * T_ + t) * D_ + d;
    s = lam[idx] * s + Bu[idx];
    Sb[idx] = (bf16_t)s;
  }
}

// ---------------- y = gelu(S@Cw)*sigmoid(...); tmp = 2*h_attn + y ----------------
__global__ __launch_bounds__(256) void combine_kernel(
    const float* __restrict__ yc, const float* __restrict__ gg,
    const float* __restrict__ ha, bf16_t* __restrict__ tmpb)
{
  size_t i = (size_t)blockIdx.x * 256 + threadIdx.x;
  float y = yc[i] * gg[i];
  tmpb[i] = (bf16_t)(2.0f * ha[i] + y);
}

// ---------------- host side ----------------
static void gemm(hipStream_t st, const bf16_t* A, const bf16_t* Wt,
                 const float* bias, const float* resid,
                 float* C, bf16_t* Cb, int M, int N, int K, int Nw, int act,
                 bool nguard)
{
  dim3 g((M + 127) / 128, (N + 63) / 64);
  int kslab = IMIN(K, 256);
  int nslab = K / kslab;
  size_t smem = (size_t)(nslab > 1 ? 2 : 1) * 64 * kslab * 2;
  if (nguard)
    gemm_bf16_kernel<true><<<g, 256, smem, st>>>(A, Wt, bias, resid, C, Cb, M, N, K, Nw, act, kslab);
  else
    gemm_bf16_kernel<false><<<g, 256, smem, st>>>(A, Wt, bias, resid, C, Cb, M, N, K, Nw, act, kslab);
}

extern "C" void kernel_launch(void* const* d_in, const int* in_sizes, int n_in,
                              void* d_out, int out_size, void* d_ws, size_t ws_size,
                              hipStream_t stream) {
  const float* x      = (const float*)d_in[0];
  const float* g1     = (const float*)d_in[1];
  const float* b1     = (const float*)d_in[2];
  const float* Wqkv   = (const float*)d_in[3];
  const float* Aqkv   = (const float*)d_in[4];
  const float* Bqkv   = (const float*)d_in[5];
  const float* Wo     = (const float*)d_in[6];
  const float* Ao     = (const float*)d_in[7];
  const float* Bo     = (const float*)d_in[8];
  const float* a_base = (const float*)d_in[9];
  const float* a_delta= (const float*)d_in[10];
  const float* Vw     = (const float*)d_in[11];
  const float* Uw     = (const float*)d_in[12];
  const float* E1w    = (const float*)d_in[13];
  const float* E1b    = (const float*)d_in[14];
  const float* E2w    = (const float*)d_in[15];
  const float* E2b    = (const float*)d_in[16];
  const float* Cw     = (const float*)d_in[17];
  const float* Gw     = (const float*)d_in[18];
  const float* Gb     = (const float*)d_in[19];
  const float* Mixw   = (const float*)d_in[20];
  const float* Mixb   = (const float*)d_in[21];
  const float* g2     = (const float*)d_in[22];
  const float* b2     = (const float*)d_in[23];
  const float* W1     = (const float*)d_in[24];
  const float* b1m    = (const float*)d_in[25];
  const float* A1l    = (const float*)d_in[26];
  const float* B1l    = (const float*)d_in[27];
  const float* W2     = (const float*)d_in[28];
  const float* b2m    = (const float*)d_in[29];
  const float* A2l    = (const float*)d_in[30];
  const float* B2l    = (const float*)d_in[31];

  char* wsp = (char*)d_ws;
  size_t off = 0;
  auto alloc = [&](size_t bytes) -> void* {
    void* p = wsp + off;
    off = (off + bytes + 255) & ~(size_t)255;
    return p;
  };

  bf16_t* Wqkv_e = (bf16_t*)alloc((size_t)MD_ * D_ * 2);
  bf16_t* Wo_e   = (bf16_t*)alloc((size_t)D_ * D_ * 2);
  bf16_t* W1_e   = (bf16_t*)alloc((size_t)MD_ * D_ * 2);
  bf16_t* W2_e   = (bf16_t*)alloc((size_t)D_ * MD_ * 2);
  bf16_t* Vw_b   = (bf16_t*)alloc((size_t)RANK_ * D_ * 2);
  bf16_t* Uw_b   = (bf16_t*)alloc((size_t)D_ * 32 * 2);     // [256,32], cols 16..31 = 0
  bf16_t* E1w_b  = (bf16_t*)alloc((size_t)D3_ * D_ * 2);
  bf16_t* Cw_b   = (bf16_t*)alloc((size_t)D_ * D_ * 2);
  bf16_t* Gw_b   = (bf16_t*)alloc((size_t)D_ * D_ * 2);
  bf16_t* Mix_b  = (bf16_t*)alloc((size_t)D_ * D_ * 2);
  bf16_t* h_bf   = (bf16_t*)alloc((size_t)BT_ * D_ * 2);
  float*  qkvf   = (float*) alloc((size_t)BT_ * 3 * D_ * 4);
  bf16_t* Qb     = (bf16_t*)alloc((size_t)B_ * H_ * T_ * DH_ * 2);
  bf16_t* Kb     = (bf16_t*)alloc((size_t)B_ * H_ * T_ * DH_ * 2);
  bf16_t* Vt     = (bf16_t*)alloc((size_t)B_ * H_ * DH_ * T_ * 2);
  bf16_t* zb     = (bf16_t*)alloc((size_t)BT_ * D_ * 2);
  float*  ha     = (float*) alloc((size_t)BT_ * D_ * 4);
  bf16_t* hab    = (bf16_t*)alloc((size_t)BT_ * D_ * 2);
  bf16_t* t1b    = (bf16_t*)alloc((size_t)BT_ * 32 * 2);    // [BT,32], cols 16..31 = 0
  float*  Bu     = (float*) alloc((size_t)BT_ * D_ * 4);
  float*  e1f    = (float*) alloc((size_t)BT_ * D3_ * 4);
  float*  lam    = (float*) alloc((size_t)BT_ * D_ * 4);
  bf16_t* Sb     = (bf16_t*)alloc((size_t)BT_ * D_ * 2);
  float*  yc     = (float*) alloc((size_t)BT_ * D_ * 4);
  float*  gg     = (float*) alloc((size_t)BT_ * D_ * 4);
  bf16_t* tmpb   = (bf16_t*)alloc((size_t)BT_ * D_ * 2);
  float*  x2     = (float*) alloc((size_t)BT_ * D_ * 4);
  bf16_t* h2b    = (bf16_t*)alloc((size_t)BT_ * D_ * 2);
  bf16_t* ub     = (bf16_t*)alloc((size_t)BT_ * MD_ * 2);

  // 1) fold LoRA into weights; convert plain weights to bf16.
  fuse_lora_kernel<<<(MD_ * D_ + 255) / 256, 256, 0, stream>>>(Wqkv, Aqkv, Bqkv, Wqkv_e, MD_, D_);
  fuse_lora_kernel<<<(D_ * D_ + 255) / 256, 256, 0, stream>>>(Wo, Ao, Bo, Wo_e, D_, D_);
  fuse_lora_kernel<<<(MD_ * D_ + 255) / 256, 256, 0, stream>>>(W1, A1l, B1l, W1_e, MD_, D_);
  fuse_lora_kernel<<<(D_ * MD_ + 255) / 256, 256, 0, stream>>>(W2, A2l, B2l, W2_e, D_, MD_);
  cvt_bf16_kernel<<<(RANK_ * D_ + 255) / 256, 256, 0, stream>>>(Vw, Vw_b, RANK_ * D_);
  cvt_pad_kernel<<<(D_ * 32 + 255) / 256, 256, 0, stream>>>(Uw, Uw_b, D_, RANK_, 32);
  cvt_bf16_kernel<<<(D3_ * D_ + 255) / 256, 256, 0, stream>>>(E1w, E1w_b, D3_ * D_);
  cvt_bf16_kernel<<<(D_ * D_ + 255) / 256, 256, 0, stream>>>(Cw, Cw_b, D_ * D_);
  cvt_bf16_kernel<<<(D_ * D_ + 255) / 256, 256, 0, stream>>>(Gw, Gw_b, D_ * D_);
  cvt_bf16_kernel<<<(D_ * D_ + 255) / 256, 256, 0, stream>>>(Mixw, Mix_b, D_ * D_);

  // 2) LN1 -> h (bf16)
  ln_kernel<<<BT_, 256, 0, stream>>>(x, g1, b1, h_bf);

  // 3) qkv = h @ Wqkv_eff^T
  gemm(stream, h_bf, Wqkv_e, nullptr, nullptr, qkvf, nullptr, BT_, MD_, D_, MD_, 0, false);

  // 4) RoPE + head split + V transpose
  rope_kernel<<<(B_ * T_ * H_ * 16) / 256, 256, 0, stream>>>(qkvf, Qb, Kb, Vt);

  // 5) causal flash attention -> z (bf16, [B,T,D] head-interleaved)
  {
    dim3 ga(T_ / 64, B_ * H_);
    attn_kernel<<<ga, 128, 0, stream>>>(Qb, Kb, Vt, zb);
  }

  // 6) h_attn = z @ Wo_eff^T (f32 + bf16)
  gemm(stream, zb, Wo_e, nullptr, nullptr, ha, hab, BT_, D_, D_, D_, 0, false);

  // 7) Bu = (h_attn @ Vw^T) @ Uw^T  -- t1 zero-padded to K=32 by TDM OOB
  gemm(stream, hab, Vw_b, nullptr, nullptr, nullptr, t1b, BT_, 32, D_, RANK_, 0, true);
  gemm(stream, t1b, Uw_b, nullptr, nullptr, Bu, nullptr, BT_, D_, 32, D_, 0, false);

  // 8) e1 = gelu(h_attn @ E1w^T + E1b); alpha, lambda
  gemm(stream, hab, E1w_b, E1b, nullptr, e1f, nullptr, BT_, D3_, D_, D3_, 1, true);
  alpha_lam_kernel<<<BT_, 256, 0, stream>>>(e1f, E2w, E2b, a_base, a_delta, lam);

  // 9) scan: S[t] = lam[t]*S[t-1] + Bu[t]
  scan_kernel<<<(B_ * D_) / 256, 256, 0, stream>>>(lam, Bu, Sb);

  // 10) y = gelu(S@Cw^T) * sigmoid(h_attn@Gw^T + Gb); tmp = 2*h_attn + y
  gemm(stream, Sb, Cw_b, nullptr, nullptr, yc, nullptr, BT_, D_, D_, D_, 1, false);
  gemm(stream, hab, Gw_b, Gb, nullptr, gg, nullptr, BT_, D_, D_, D_, 2, false);
  combine_kernel<<<(BT_ * D_) / 256, 256, 0, stream>>>(yc, gg, ha, tmpb);

  // 11) x2 = x + tmp @ Mixw^T + Mixb
  gemm(stream, tmpb, Mix_b, Mixb, x, x2, nullptr, BT_, D_, D_, D_, 0, false);

  // 12) LN2 -> h2 (bf16)
  ln_kernel<<<BT_, 256, 0, stream>>>(x2, g2, b2, h2b);

  // 13) MLP: u = gelu(h2@W1_eff^T + b1m); out = x2 + u@W2_eff^T + b2m
  gemm(stream, h2b, W1_e, b1m, nullptr, nullptr, ub, BT_, MD_, D_, MD_, 1, false);
  gemm(stream, ub, W2_e, b2m, x2, (float*)d_out, nullptr, BT_, D_, MD_, D_, 0, false);
}
